// Attention_53420803228253
// MI455X (gfx1250) — matrix operand order repaired and checked by reading, not compiled, not run on hardware
//
#include <hip/hip_runtime.h>
#include <hip/hip_bf16.h>

// ---------------------------------------------------------------------------
// Fused attention (fwd, returns output AND attention matrix) for
// B=8, H=8, LQ=LK=1024, D=64, fp32.
// Memory-bound: mask read (256MB) + attention write (256MB) dominate.
// Compute: v_wmma_f32_16x16x32_f16 (fp32 accumulate).
// Attention tile is written with GLOBAL_STORE_ASYNC_FROM_LDS_B128 so the
// dominant store stream overlaps the P*V WMMA phase (ASYNCcnt drains at
// s_endpgm's implicit wait-idle).
// ---------------------------------------------------------------------------

typedef _Float16 half_t;
typedef __attribute__((ext_vector_type(16))) half_t v16h;
typedef __attribute__((ext_vector_type(8)))  half_t v8h;
typedef __attribute__((ext_vector_type(8)))  float  v8f;
typedef __attribute__((ext_vector_type(4)))  float  v4f;
typedef __attribute__((ext_vector_type(4)))  int    v4i;

// async-to/from-LDS builtins take int4* in global (as1) / LDS (as3) space
typedef __attribute__((address_space(1))) v4i* as1_v4i_ptr;
typedef __attribute__((address_space(3))) v4i* as3_v4i_ptr;

#if defined(__has_builtin)
#if __has_builtin(__builtin_amdgcn_global_store_async_from_lds_b128)
#define USE_ASYNC_STORE 1
#endif
#if __has_builtin(__builtin_amdgcn_global_load_async_to_lds_b128) && \
    __has_builtin(__builtin_amdgcn_s_wait_asynccnt)
#define USE_ASYNC_LOAD 1
#endif
#endif

#define BQ    32                      // query rows per workgroup
#define BK    64                      // key rows per k-tile
#define DIM   64
#define LQ    1024
#define LK    1024
#define NBH   64                      // B*H
#define LDE   1028                    // fp32 score-row stride (padded)
#define LDH   72                      // f16 row stride in halves (padded)
#define LDK   68                      // fp32 K-stage row stride (padded)
#define E_BYTES   (BQ * LDE * 4)      // 131584
#define KV_BYTES  (BK * LDK * 4)      // 17408 (fits f16 VT tile too)
#define Q_BYTES   (BQ * LDH * 2)      // 4608
#define SMEM_BYTES (E_BYTES + KV_BYTES + Q_BYTES)   // 153600 -> 2 WGs/WGP

__device__ __forceinline__ v8f wmma_f16(v16h a, v16h b, v8f c) {
    return __builtin_amdgcn_wmma_f32_16x16x32_f16(
        /*neg_a=*/false, a, /*neg_b=*/false, b,
        /*c_mod=*/(short)0, c, /*reuse_a=*/false, /*reuse_b=*/false);
}

// A operand (16x32 f16 slice): lane half selects contiguous 16 K values (K = sel*16..+15)
__device__ __forceinline__ v16h load_a_f16(const half_t* base, int row, int kcol, int lane) {
    const half_t* p = base + row * LDH + kcol;
    int sel = (lane >> 4) & 1;
    v8h lo = *(const v8h*)(p + sel * 16);
    v8h hi = *(const v8h*)(p + sel * 16 + 8);
    v16h a;
#pragma unroll
    for (int i = 0; i < 8; ++i) { a[i] = lo[i]; a[8 + i] = hi[i]; }
    return a;
}

// B operand (32x16 f16 slice): lane half selects contiguous 16 K values
__device__ __forceinline__ v16h load_b_f16(const half_t* base, int row, int kcol, int lane) {
    const half_t* p = base + row * LDH + kcol + ((lane >> 4) & 1) * 16;
    v8h lo = ((const v8h*)p)[0];
    v8h hi = ((const v8h*)p)[1];
    v16h b;
#pragma unroll
    for (int i = 0; i < 8; ++i) { b[i] = lo[i]; b[8 + i] = hi[i]; }
    return b;
}

// B operand from fp32 LDS rows (async-staged K), cvt -> f16
__device__ __forceinline__ v16h load_b_f32cvt(const float* base, int row, int kcol, int lane) {
    const float* p = base + row * LDK + kcol + ((lane >> 4) & 1) * 16;
    v4f f0 = ((const v4f*)p)[0];
    v4f f1 = ((const v4f*)p)[1];
    v4f f2 = ((const v4f*)p)[2];
    v4f f3 = ((const v4f*)p)[3];
    v16h b;
#pragma unroll
    for (int i = 0; i < 4; ++i) {
        b[i]      = (half_t)f0[i];
        b[4 + i]  = (half_t)f1[i];
        b[8 + i]  = (half_t)f2[i];
        b[12 + i] = (half_t)f3[i];
    }
    return b;
}

// A operand built on the fly from fp32 LDS (normalized P), cvt -> f16
__device__ __forceinline__ v16h load_a_f32cvt(const float* base, int row, int kcol, int lane) {
    const float* p = base + row * LDE + kcol;
    int sel = (lane >> 4) & 1;
    v4f f0 = ((const v4f*)(p + sel * 16))[0];
    v4f f1 = ((const v4f*)(p + sel * 16))[1];
    v4f f2 = ((const v4f*)(p + sel * 16))[2];
    v4f f3 = ((const v4f*)(p + sel * 16))[3];
    v16h a;
#pragma unroll
    for (int i = 0; i < 4; ++i) {
        a[i]      = (half_t)f0[i];
        a[4 + i]  = (half_t)f1[i];
        a[8 + i]  = (half_t)f2[i];
        a[12 + i] = (half_t)f3[i];
    }
    return a;
}

__global__ __launch_bounds__(256)
void attn_fused_kernel(const float* __restrict__ Qg,
                       const float* __restrict__ Kg,
                       const float* __restrict__ Vg,
                       const int*   __restrict__ maskg,
                       const float* __restrict__ qmaskg,
                       float* __restrict__ outg,
                       float* __restrict__ attg) {
    extern __shared__ char smem[];
    float*  Elds   = (float*)smem;                         // BQ x LDE fp32 (scores -> P)
    float*  Kstage = (float*)(smem + E_BYTES);             // BK x LDK fp32 (K tiles)
    half_t* VTlds  = (half_t*)(smem + E_BYTES);            // BK x LDH f16 (V^T, reuses Kstage)
    half_t* Qlds   = (half_t*)(smem + E_BYTES + KV_BYTES); // BQ x LDH f16

    const int t    = threadIdx.x;
    const int lane = t & 31;
    const int w    = t >> 5;           // wave 0..7
    const int qi   = w & 1;            // q sub-tile (16 rows)
    const int kx   = w >> 1;           // k/d sub-tile (0..3)
    const int bh   = blockIdx.y;
    const int q0   = blockIdx.x * BQ;

    const size_t qkv_base = (size_t)bh * (LQ * DIM);

    // ---------------- Phase 0: Q tile -> LDS f16 (pre-scaled by 1/8) -------
    {
        int row = t >> 3, seg = t & 7;                 // 8 floats / thread
        const float* gp = Qg + qkv_base + (size_t)(q0 + row) * DIM + seg * 8;
        v4f f0 = ((const v4f*)gp)[0];
        v4f f1 = ((const v4f*)gp)[1];
        v8h h;
#pragma unroll
        for (int i = 0; i < 4; ++i) {
            h[i]     = (half_t)(f0[i] * 0.125f);
            h[4 + i] = (half_t)(f1[i] * 0.125f);
        }
        *(v8h*)(Qlds + row * LDH + seg * 8) = h;
    }
    __syncthreads();

    // Hoist A operands (constant across k tiles)
    v16h aq0 = load_a_f16(Qlds, 16 * qi + (lane & 15), 0, lane);
    v16h aq1 = load_a_f16(Qlds, 16 * qi + (lane & 15), 32, lane);

    // ---------------- Phase 1: scores = (Q/8)K^T, mask, store to LDS -------
    const int* mrow = maskg + (size_t)bh * LQ * LK + (size_t)q0 * LK;
    for (int kt = 0; kt < LK / BK; ++kt) {
        __syncthreads();
#if defined(USE_ASYNC_LOAD)
        {   // stage K tile (fp32) via async global->LDS DMA
            const float* gp0 = Kg + qkv_base + (size_t)(kt * BK) * DIM;
#pragma unroll
            for (int i = 0; i < 4; ++i) {
                int linear = i * 256 + t;              // 1024 b128 chunks
                int row = linear >> 4;                 // 0..63
                int col = (linear & 15) * 4;           // float column
                __builtin_amdgcn_global_load_async_to_lds_b128(
                    (as1_v4i_ptr)(gp0 + (size_t)row * DIM + col),
                    (as3_v4i_ptr)(Kstage + row * LDK + col),
                    0, 0);
            }
            __builtin_amdgcn_s_wait_asynccnt(0);
        }
#else
        {   // stage K tile -> f16 LDS through VGPRs
            int row = t >> 2, seg = t & 3;             // 16 floats / thread
            const float* gp = Kg + qkv_base + (size_t)(kt * BK + row) * DIM + seg * 16;
            v4f f0 = ((const v4f*)gp)[0];
            v4f f1 = ((const v4f*)gp)[1];
            v4f f2 = ((const v4f*)gp)[2];
            v4f f3 = ((const v4f*)gp)[3];
            v8h h0, h1;
#pragma unroll
            for (int i = 0; i < 4; ++i) {
                h0[i] = (half_t)f0[i]; h0[4 + i] = (half_t)f1[i];
                h1[i] = (half_t)f2[i]; h1[4 + i] = (half_t)f3[i];
            }
            half_t* kh = (half_t*)Kstage;
            v8h* dp = (v8h*)(kh + row * LDH + seg * 16);
            dp[0] = h0; dp[1] = h1;
        }
#endif
        __syncthreads();

#if defined(USE_ASYNC_LOAD)
        v16h b0 = load_b_f32cvt(Kstage, 16 * kx + (lane & 15), 0, lane);
        v16h b1 = load_b_f32cvt(Kstage, 16 * kx + (lane & 15), 32, lane);
#else
        v16h b0 = load_b_f16((const half_t*)Kstage, 16 * kx + (lane & 15), 0, lane);
        v16h b1 = load_b_f16((const half_t*)Kstage, 16 * kx + (lane & 15), 32, lane);
#endif
        v8f c = {0.f, 0.f, 0.f, 0.f, 0.f, 0.f, 0.f, 0.f};
        c = wmma_f16(aq0, b0, c);
        c = wmma_f16(aq1, b1, c);

        // apply mask, store fp32 scores to LDS
        int n = lane & 15, hi = lane >> 4;
        int kcol = kt * BK + 16 * kx + n;
#pragma unroll
        for (int g = 0; g < 8; ++g) {
            int qr = 16 * qi + g + 8 * hi;
            int mv = __builtin_nontemporal_load(mrow + (size_t)qr * LK + kcol);
            Elds[qr * LDE + kcol] = mv ? c[g] : -1.0e9f;
        }
    }
    __syncthreads();

    // ---------- Phase 2: softmax in LDS, normalized IN PLACE (P = E*qm/l) --
    for (int r = w * 4; r < w * 4 + 4; ++r) {
        float* er = Elds + r * LDE;
        float m = -3.0e38f;
#pragma unroll 4
        for (int j = 0; j < LK / 32; ++j) m = fmaxf(m, er[j * 32 + lane]);
#pragma unroll
        for (int off = 16; off; off >>= 1) m = fmaxf(m, __shfl_xor(m, off, 32));
        float l = 0.f;
#pragma unroll 4
        for (int j = 0; j < LK / 32; ++j) {
            float e = __expf(er[j * 32 + lane] - m);
            er[j * 32 + lane] = e;
            l += e;
        }
#pragma unroll
        for (int off = 16; off; off >>= 1) l += __shfl_xor(l, off, 32);
        float sc = qmaskg[(size_t)bh * LQ + q0 + r] / l;   // uniform in wave
#pragma unroll 4
        for (int j = 0; j < LK / 32; ++j) er[j * 32 + lane] *= sc;
    }
    __syncthreads();

    // ---------- Phase 3: stream attention tile (pure LDS -> global copy) ---
    {
        float* attb = attg + (size_t)bh * LQ * LK + (size_t)q0 * LK;
#if defined(USE_ASYNC_STORE)
#pragma unroll 4
        for (int i = 0; i < 32; ++i) {                 // 32 b128 chunks / thread
            int linear = i * 256 + t;
            int r = linear >> 8, c4 = linear & 255;
            __builtin_amdgcn_global_store_async_from_lds_b128(
                (as1_v4i_ptr)(attb + (size_t)r * LK + c4 * 4),
                (as3_v4i_ptr)(Elds + r * LDE + c4 * 4),
                0, 0);
        }
        // No wait: Elds is only READ below; ASYNCcnt drains at s_endpgm.
#else
#pragma unroll 4
        for (int i = 0; i < 32; ++i) {
            int linear = i * 256 + t;
            int r = linear >> 8, c4 = linear & 255;
            v4f v = *(const v4f*)(Elds + r * LDE + c4 * 4);
            __builtin_nontemporal_store(v, (v4f*)(attb + (size_t)r * LK + c4 * 4));
        }
#endif
    }

    // ---------------- Phase 4: O = P * V (overlaps async att stores) -------
    v8f co = {0.f, 0.f, 0.f, 0.f, 0.f, 0.f, 0.f, 0.f};
    for (int kt = 0; kt < LK / BK; ++kt) {
        __syncthreads();
        {   // stage V tile transposed -> f16 LDS [d][k]
            int row = t >> 2, seg = t & 3;
            const float* gp = Vg + qkv_base + (size_t)(kt * BK + row) * DIM + seg * 16;
            v4f f[4];
            f[0] = ((const v4f*)gp)[0];
            f[1] = ((const v4f*)gp)[1];
            f[2] = ((const v4f*)gp)[2];
            f[3] = ((const v4f*)gp)[3];
#pragma unroll
            for (int i = 0; i < 16; ++i) {
                VTlds[(seg * 16 + i) * LDH + row] = (half_t)f[i >> 2][i & 3];
            }
        }
        __syncthreads();
#pragma unroll
        for (int kc = 0; kc < 2; ++kc) {
            v16h ap = load_a_f32cvt(Elds, 16 * qi + (lane & 15), kt * BK + kc * 32, lane);
            v16h bv = load_b_f16(VTlds, 16 * kx + (lane & 15), kc * 32, lane);
            co = wmma_f16(ap, bv, co);
        }
    }

    // store output (P already carries qmask/l normalization)
    {
        int n = lane & 15, hi = lane >> 4;
        int d = 16 * kx + n;
        float* ob = outg + (size_t)bh * LQ * DIM + (size_t)q0 * DIM;
#pragma unroll
        for (int g = 0; g < 8; ++g) {
            int qr = 16 * qi + g + 8 * hi;
            ob[(size_t)qr * DIM + d] = co[g];
        }
    }

#if defined(USE_ASYNC_STORE) && defined(USE_ASYNC_LOAD)
    __builtin_amdgcn_s_wait_asynccnt(0);   // explicit drain (endpgm would too)
#endif
}

extern "C" void kernel_launch(void* const* d_in, const int* in_sizes, int n_in,
                              void* d_out, int out_size, void* d_ws, size_t ws_size,
                              hipStream_t stream) {
    (void)in_sizes; (void)n_in; (void)out_size; (void)d_ws; (void)ws_size;
    const float* Q     = (const float*)d_in[0];
    const float* K     = (const float*)d_in[1];
    const float* V     = (const float*)d_in[2];
    const int*   mask  = (const int*)d_in[3];
    const float* qmask = (const float*)d_in[4];

    float* out = (float*)d_out;                         // [8,8,1024,64]
    float* att = out + (size_t)NBH * LQ * DIM;          // [8,8,1024,1024]

    (void)hipFuncSetAttribute((const void*)attn_fused_kernel,
                              hipFuncAttributeMaxDynamicSharedMemorySize,
                              SMEM_BYTES);

    dim3 grid(LQ / BQ, NBH);   // 32 x 64
    dim3 block(256);
    attn_fused_kernel<<<grid, block, SMEM_BYTES, stream>>>(Q, K, V, mask, qmask, out, att);
}
